// Attention_1752346657109
// MI455X (gfx1250) — compile-verified, matrix operand order checked by reading, not run on hardware
//
#include <hip/hip_runtime.h>
#include <math.h>

typedef __attribute__((ext_vector_type(16))) __bf16 v16bf;
typedef __attribute__((ext_vector_type(8)))  float  v8f;
typedef __attribute__((ext_vector_type(4)))  unsigned int v4u;
typedef __attribute__((ext_vector_type(8)))  int v8i;
typedef __attribute__((ext_vector_type(4)))  int v4i;
typedef unsigned short u16;
typedef unsigned int   u32;

// ---------- bf16 helpers (round-to-nearest-even) ----------
__device__ __forceinline__ u16 f2bf(float f) {
  u32 u = __builtin_bit_cast(u32, f);
  u32 r = u + 0x7FFFu + ((u >> 16) & 1u);
  return (u16)(r >> 16);
}
__device__ __forceinline__ float bf2f(u16 h) {
  return __builtin_bit_cast(float, (u32)h << 16);
}
__device__ __forceinline__ u32 pack2(float a, float b) {
  return (u32)f2bf(a) | ((u32)f2bf(b) << 16);
}

// ---------- fragment load: 16 bf16 per lane as 2x b128 ----------
union FragU { uint4 u[2]; v16bf v; };

__device__ __forceinline__ v16bf ld_frag(const u16* rowbase, int koff0) {
  FragU r;
  r.u[0] = *reinterpret_cast<const uint4*>(rowbase + koff0);
  r.u[1] = *reinterpret_cast<const uint4*>(rowbase + koff0 + 16);
  return r.v;
}

// ---------- TDM: 2-D tensor tile DMA, global -> LDS (bf16 elements) ----
// D# per CDNA5 ISA: group0 {count, lds_addr,
// global_addr[56:0], type=2}; group1 {data_size=2B, tensor dims, tile dims,
// dim0 stride}; remaining groups zero (<=2-D tensor).
// clang-23 / therock toolchain: 6-arg builtin (extra int32x8 group).
__device__ __forceinline__ void tdm_load_2d(unsigned lds_off, const void* gaddr,
                                            unsigned tile_d0, unsigned tile_d1,
                                            unsigned tensor_d0, unsigned tensor_d1,
                                            unsigned stride_d0_elems) {
  unsigned long long ga = (unsigned long long)(size_t)gaddr;
  v4u g0;
  g0[0] = 1u;                                        // count=1 (user mode)
  g0[1] = lds_off;                                   // lds_addr (bytes)
  g0[2] = (u32)(ga & 0xFFFFFFFFu);                   // global_addr[31:0]
  g0[3] = (u32)((ga >> 32) & 0x01FFFFFFu) | (2u << 30);  // addr[56:32] | type=2
  v8i g1;
  g1[0] = (int)(1u << 16);                           // data_size=1 -> 2 bytes
  g1[1] = (int)((tensor_d0 & 0xFFFFu) << 16);        // tensor_dim0[15:0]
  g1[2] = (int)(((tensor_d0 >> 16) & 0xFFFFu) | ((tensor_d1 & 0xFFFFu) << 16));
  g1[3] = (int)(((tensor_d1 >> 16) & 0xFFFFu) | ((tile_d0 & 0xFFFFu) << 16));
  g1[4] = (int)(tile_d1 & 0xFFFFu);                  // tile_dim1 | tile_dim2=0
  g1[5] = (int)stride_d0_elems;                      // tensor_dim0_stride[31:0]
  g1[6] = 0;
  g1[7] = 0;
  v4i z4 = {0, 0, 0, 0};
  v8i z8 = {0, 0, 0, 0, 0, 0, 0, 0};
  __builtin_amdgcn_tensor_load_to_lds(g0, g1, z4, z4, z8, 0);
}

// =====================================================================
// GEMM: C[M,N] = A[M,K] * W[N,K]^T.  A is f32 or bf16; C is bf16 or f32.
// Block tile 128x128, K-step 32, 8 waves each computing a 32x64 tile.
// f32 operands converted to bf16 while staging into LDS.
// =====================================================================
template <bool A_BF16, bool OUT_F32>
__global__ void __launch_bounds__(256)
gemm_bf16_wmma(const void* __restrict__ Aptr, const float* __restrict__ W,
               void* __restrict__ Cptr, int M, int N, int K) {
  __shared__ u16 As[128 * 32];   // [m][k]
  __shared__ u16 Bs[128 * 32];   // [n][k]

  const int tid   = threadIdx.x;
  const int lane  = tid & 31;
  const int wave  = tid >> 5;
  const int wr    = wave >> 1;          // 0..3 -> M offset wr*32
  const int wc    = wave & 1;           // 0..1 -> N offset wc*64
  const int blockN = blockIdx.x * 128;
  const int blockM = blockIdx.y * 128;

  const int lrow = tid >> 1;            // 0..127 (staging row)
  const int lcol = (tid & 1) * 16;      // 0 or 16 (staging col)

  const int mloc  = lane & 15;          // row/col within a 16-wide frag
  const int koff0 = (lane >> 4) * 8;    // K half select per ISA A-layout

  v8f acc[2][4] = {};

  for (int k0 = 0; k0 < K; k0 += 32) {
    __syncthreads();
    // ---- stage A tile 128x32 as bf16 ----
    if constexpr (A_BF16) {
      const u16* Ab = (const u16*)Aptr;
      const uint4* src =
          reinterpret_cast<const uint4*>(Ab + (size_t)(blockM + lrow) * K + k0 + lcol);
      *reinterpret_cast<uint4*>(&As[lrow * 32 + lcol])     = src[0];
      *reinterpret_cast<uint4*>(&As[lrow * 32 + lcol + 8]) = src[1];
    } else {
      const float* Af = (const float*)Aptr;
      const float4* src =
          reinterpret_cast<const float4*>(Af + (size_t)(blockM + lrow) * K + k0 + lcol);
      float4 f0 = src[0], f1 = src[1], f2 = src[2], f3 = src[3];
      uint4 p0, p1;
      p0.x = pack2(f0.x, f0.y); p0.y = pack2(f0.z, f0.w);
      p0.z = pack2(f1.x, f1.y); p0.w = pack2(f1.z, f1.w);
      p1.x = pack2(f2.x, f2.y); p1.y = pack2(f2.z, f2.w);
      p1.z = pack2(f3.x, f3.y); p1.w = pack2(f3.z, f3.w);
      *reinterpret_cast<uint4*>(&As[lrow * 32 + lcol])     = p0;
      *reinterpret_cast<uint4*>(&As[lrow * 32 + lcol + 8]) = p1;
    }
    // ---- stage W tile 128x32 (always f32 in memory) ----
    {
      const float4* src =
          reinterpret_cast<const float4*>(W + (size_t)(blockN + lrow) * K + k0 + lcol);
      float4 f0 = src[0], f1 = src[1], f2 = src[2], f3 = src[3];
      uint4 p0, p1;
      p0.x = pack2(f0.x, f0.y); p0.y = pack2(f0.z, f0.w);
      p0.z = pack2(f1.x, f1.y); p0.w = pack2(f1.z, f1.w);
      p1.x = pack2(f2.x, f2.y); p1.y = pack2(f2.z, f2.w);
      p1.z = pack2(f3.x, f3.y); p1.w = pack2(f3.z, f3.w);
      *reinterpret_cast<uint4*>(&Bs[lrow * 32 + lcol])     = p0;
      *reinterpret_cast<uint4*>(&Bs[lrow * 32 + lcol + 8]) = p1;
      if (k0 + 32 < K)  // global_prefetch_b8 on the next K-slab of weights
        __builtin_prefetch(W + (size_t)(blockN + lrow) * K + k0 + 32 + lcol, 0, 1);
    }
    __syncthreads();

    v16bf afr[2], bfr[4];
#pragma unroll
    for (int i = 0; i < 2; ++i)
      afr[i] = ld_frag(&As[(wr * 32 + i * 16 + mloc) * 32], koff0);
#pragma unroll
    for (int j = 0; j < 4; ++j)
      bfr[j] = ld_frag(&Bs[(wc * 64 + j * 16 + mloc) * 32], koff0);

#pragma unroll
    for (int i = 0; i < 2; ++i)
#pragma unroll
      for (int j = 0; j < 4; ++j)
        acc[i][j] = __builtin_amdgcn_wmma_f32_16x16x32_bf16(
            false, afr[i], false, bfr[j], (short)0, acc[i][j], false, false);
  }

  // ---- epilogue: C-frag layout (lane<16: M=e, lane>=16: M=e+8; N=lane&15) ----
#pragma unroll
  for (int i = 0; i < 2; ++i) {
#pragma unroll
    for (int j = 0; j < 4; ++j) {
      const int col  = blockN + wc * 64 + j * 16 + mloc;
      const int row0 = blockM + wr * 32 + i * 16 + (lane >> 4) * 8;
#pragma unroll
      for (int e = 0; e < 8; ++e) {
        const int row = row0 + e;
        if constexpr (OUT_F32)
          ((float*)Cptr)[(size_t)row * N + col] = acc[i][j][e];
        else
          ((u16*)Cptr)[(size_t)row * N + col] = f2bf(acc[i][j][e]);
      }
    }
  }
}

// =====================================================================
// RoPE on bf16 [S][H][128] in place, interleaved (even,odd) pairs.
// =====================================================================
__global__ void rope_kernel(u16* __restrict__ t, const float* __restrict__ cosv,
                            const float* __restrict__ sinv, int H) {
  int idx = blockIdx.x * blockDim.x + threadIdx.x;
  int total = 4096 * H * 64;
  if (idx >= total) return;
  int d = idx & 63;
  int h = (idx >> 6) % H;
  int s = idx / (64 * H);
  size_t base = ((size_t)s * H + h) * 128 + 2 * d;
  float a = bf2f(t[base]), b = bf2f(t[base + 1]);
  float c = cosv[s * 64 + d], sn = sinv[s * 64 + d];
  t[base]     = f2bf(a * c - b * sn);
  t[base + 1] = f2bf(a * sn + b * c);
}

// =====================================================================
// Flash attention: grid(64 q-tiles, 32 heads), 128 threads = 4 waves.
// Each wave owns 16 query rows; loop over 32-key chunks with online
// softmax.  K chunk DMA'd into LDS by the Tensor Data Mover (one issue
// per block, overlapped with the ALU-side V-transpose staging); V chunk
// staged transposed; P re-laid-out via wave-private LDS.
// __launch_bounds__(128, 1): trade occupancy for registers so all B
// fragments stay resident and ds_load_b128s pipeline ahead of WMMAs.
// =====================================================================
__global__ void __launch_bounds__(128, 1)
attn_kernel(const u16* __restrict__ q, const u16* __restrict__ k,
            const u16* __restrict__ v, u16* __restrict__ o) {
  // single LDS block with fixed layout (byte offsets needed for the TDM D#)
  __shared__ __align__(16) u16 smem[32 * 128 + 128 * 32 + 4 * 16 * 32];
  u16* Ks = smem;                       // K chunk [32 keys][128 dims], 8 KB
  u16* Vt = smem + 32 * 128;            // V chunk transposed [d][kv],   8 KB
  u16* Pstage = smem + 32 * 128 + 128 * 32;  // 4 wave-private P tiles,  4 KB

  const int h    = blockIdx.y;
  const int kvh  = h >> 2;            // GQA: 4 Q heads per KV head
  const int tid  = threadIdx.x;
  const int wave = tid >> 5;
  const int lane = tid & 31;
  const int mloc  = lane & 15;
  const int koff0 = (lane >> 4) * 8;
  const int qrow0 = blockIdx.x * 64 + wave * 16;

  // Q fragments: 16 x 128 = 4 A-frags of 16x32
  v16bf qa[4];
#pragma unroll
  for (int i = 0; i < 4; ++i)
    qa[i] = ld_frag(q + (size_t)(qrow0 + mloc) * 4096 + h * 128 + i * 32, koff0);

  v8f   O[8] = {};
  float mrow[8], lsum[8], alv[8];
#pragma unroll
  for (int e = 0; e < 8; ++e) { mrow[e] = -3.0e38f; lsum[e] = 0.f; }

  const float scale = 0.08838834764831845f;  // 1/sqrt(128)
  u16* Pst = Pstage + wave * (16 * 32);
  const unsigned ldsK = (unsigned)(size_t)(void*)Ks;  // LDS byte offset

  for (int t0 = 0; t0 < 4096; t0 += 32) {
    __syncthreads();
    // ---- TDM: DMA K chunk (32 keys x 128 dims, bf16) into LDS.  One
    // issue per block (wave 0); DMA overlaps the V staging below. ----
    if (wave == 0) {
      const u16* gk = k + (size_t)t0 * 1024 + kvh * 128;
      tdm_load_2d(ldsK, gk, /*tile_d0=*/128, /*tile_d1=*/32,
                  /*tensor_d0=*/128, /*tensor_d1=*/(unsigned)(4096 - t0),
                  /*stride_d0_elems=*/1024);
    }
    // ---- cooperative V chunk stage (32 keys x 128 dims), transposed ----
    {
      const int kr = tid >> 2;          // key row 0..31
      const int cb = (tid & 3) * 32;    // dim col base
      const u16* vb = v + (size_t)(t0 + kr) * 1024 + kvh * 128 + cb;
#pragma unroll
      for (int j4 = 0; j4 < 4; ++j4) {
        uint4 g = *reinterpret_cast<const uint4*>(vb + j4 * 8);
        u32 w4[4] = {g.x, g.y, g.z, g.w};
#pragma unroll
        for (int p = 0; p < 4; ++p) {
          int c0 = cb + j4 * 8 + p * 2;
          Vt[(c0 + 0) * 32 + kr] = (u16)w4[p];
          Vt[(c0 + 1) * 32 + kr] = (u16)(w4[p] >> 16);
        }
      }
    }
    if (wave == 0) __builtin_amdgcn_s_wait_tensorcnt(0);
    __syncthreads();

    // ---- preload all 8 K fragments, then 8 back-to-back WMMAs ----
    v16bf bk[2][4];
#pragma unroll
    for (int tt = 0; tt < 2; ++tt)
#pragma unroll
      for (int i = 0; i < 4; ++i)
        bk[tt][i] = ld_frag(&Ks[(tt * 16 + mloc) * 128 + i * 32], koff0);

    v8f S[2];
#pragma unroll
    for (int tt = 0; tt < 2; ++tt) {
      v8f s = {};
#pragma unroll
      for (int i = 0; i < 4; ++i)
        s = __builtin_amdgcn_wmma_f32_16x16x32_bf16(false, qa[i], false, bk[tt][i],
                                                    (short)0, s, false, false);
#pragma unroll
      for (int e = 0; e < 8; ++e) s[e] *= scale;
      S[tt] = s;
    }

    // ---- online softmax: row reductions across the 16-lane half ----
#pragma unroll
    for (int e = 0; e < 8; ++e) {
      float x = fmaxf(S[0][e], S[1][e]);
      x = fmaxf(x, __shfl_xor(x, 1, 32));
      x = fmaxf(x, __shfl_xor(x, 2, 32));
      x = fmaxf(x, __shfl_xor(x, 4, 32));
      x = fmaxf(x, __shfl_xor(x, 8, 32));
      float mn = fmaxf(mrow[e], x);
      alv[e]  = __expf(mrow[e] - mn);
      mrow[e] = mn;
    }
#pragma unroll
    for (int e = 0; e < 8; ++e) {
      float p0 = __expf(S[0][e] - mrow[e]);
      float p1 = __expf(S[1][e] - mrow[e]);
      S[0][e] = p0; S[1][e] = p1;
      float r = p0 + p1;
      r += __shfl_xor(r, 1, 32);
      r += __shfl_xor(r, 2, 32);
      r += __shfl_xor(r, 4, 32);
      r += __shfl_xor(r, 8, 32);
      lsum[e] = lsum[e] * alv[e] + r;
    }
#pragma unroll
    for (int j = 0; j < 8; ++j)
#pragma unroll
      for (int e = 0; e < 8; ++e) O[j][e] *= alv[e];

    // ---- P: C-frag layout -> bf16 -> LDS -> A-frag layout ----
#pragma unroll
    for (int tt = 0; tt < 2; ++tt)
#pragma unroll
      for (int e = 0; e < 8; ++e) {
        int r = e + (lane >> 4) * 8;
        Pst[r * 32 + tt * 16 + mloc] = f2bf(S[tt][e]);
      }

    // ---- preload all 8 V fragments + P fragment, then 8 WMMAs ----
    v16bf bv[8];
#pragma unroll
    for (int j = 0; j < 8; ++j)
      bv[j] = ld_frag(&Vt[(j * 16 + mloc) * 32], koff0);
    v16bf pa = ld_frag(&Pst[mloc * 32], koff0);

#pragma unroll
    for (int j = 0; j < 8; ++j)
      O[j] = __builtin_amdgcn_wmma_f32_16x16x32_bf16(false, pa, false, bv[j],
                                                     (short)0, O[j], false, false);
  }

  // ---- normalize and write bf16 attn output [S][32*128] ----
#pragma unroll
  for (int e = 0; e < 8; ++e) {
    float inv = 1.0f / lsum[e];
    int row = qrow0 + e + (lane >> 4) * 8;
#pragma unroll
    for (int j = 0; j < 8; ++j) {
      int col = h * 128 + j * 16 + mloc;
      o[(size_t)row * 4096 + col] = f2bf(O[j][e] * inv);
    }
  }
}

// =====================================================================
extern "C" void kernel_launch(void* const* d_in, const int* in_sizes, int n_in,
                              void* d_out, int out_size, void* d_ws, size_t ws_size,
                              hipStream_t stream) {
  const float* x    = (const float*)d_in[0];
  const float* cosv = (const float*)d_in[1];
  const float* sinv = (const float*)d_in[2];
  const float* wq   = (const float*)d_in[3];
  const float* wk   = (const float*)d_in[4];
  const float* wv   = (const float*)d_in[5];
  const float* wo   = (const float*)d_in[6];
  float* out = (float*)d_out;

  // workspace layout (bf16): q 32MB | k 8MB | v 8MB | attn 32MB  (~80MB)
  u16* qws = (u16*)d_ws;
  u16* kws = qws + (size_t)4096 * 4096;
  u16* vws = kws + (size_t)4096 * 1024;
  u16* aws = vws + (size_t)4096 * 1024;

  // QKV projections (f32 in, bf16 out)
  gemm_bf16_wmma<false, false><<<dim3(32, 32), 256, 0, stream>>>(x, wq, qws, 4096, 4096, 4096);
  gemm_bf16_wmma<false, false><<<dim3(8, 32), 256, 0, stream>>>(x, wk, kws, 4096, 1024, 4096);
  gemm_bf16_wmma<false, false><<<dim3(8, 32), 256, 0, stream>>>(x, wv, vws, 4096, 1024, 4096);

  // RoPE on Q and K
  { int total = 4096 * 32 * 64;
    rope_kernel<<<(total + 255) / 256, 256, 0, stream>>>(qws, cosv, sinv, 32); }
  { int total = 4096 * 8 * 64;
    rope_kernel<<<(total + 255) / 256, 256, 0, stream>>>(kws, cosv, sinv, 8); }

  // Flash attention (bf16 in, bf16 out), TDM-staged K tiles
  attn_kernel<<<dim3(64, 32), 128, 0, stream>>>(qws, kws, vws, aws);

  // Output projection (bf16 A, f32 out)
  gemm_bf16_wmma<true, true><<<dim3(32, 32), 256, 0, stream>>>(aws, wo, out, 4096, 4096, 4096);
}
